// LightweightConv1d_24369644438242
// MI455X (gfx1250) — matrix operand order repaired and checked by reading, not compiled, not run on hardware
//
#include <hip/hip_runtime.h>
#include <hip/hip_bf16.h>

// LightweightConv1d (B=8, C=1024, T=4096, H=16, K=31, PAD=15) for gfx1250.
// out[b,ch,t] = bias[h] + sum_m softmax(w[h])[m] * inp[b,ch,t+m-15],  h = ch/64
//
// Strategy: map the 31-tap depthwise conv onto V_WMMA_F32_16X16X32_F16.
// D(16 times x 16 channels) = A0(16x32 weight Toeplitz band) * B0(32 times x 16 ch)
//                           + A1 * B1 (next 32-time window) + bias.
// A matrices are built once per wave; the time loop is pure loads + 2 WMMA + stores.

typedef __attribute__((ext_vector_type(16))) _Float16 v16h;
typedef __attribute__((ext_vector_type(8)))  float    v8f;

#define BB   8
#define CC   1024
#define TT   4096
#define HH   16
#define KK   31
#define CPH  64          // channels per head
#define CTT  (CC * TT)   // per-batch stride

__global__ __launch_bounds__(256) void
lconv_wmma_kernel(const float* __restrict__ inp,
                  const float* __restrict__ weight,
                  const float* __restrict__ bias,
                  float* __restrict__ out) {
  const int lane      = threadIdx.x & 31;
  const int waveInBlk = threadIdx.x >> 5;
  const int W         = blockIdx.x * 8 + waveInBlk;   // 0..8191

  const int tchunk = W & 15;          // 16 chunks of 256 output times
  const int ctile  = W >> 4;          // 0..511 : (b, head, 16-ch block)
  const int b      = ctile >> 6;
  const int rem    = ctile & 63;
  const int h      = rem >> 2;
  const int cblk   = rem & 3;
  const int ch0    = h * CPH + cblk * 16;

  // ---- per-head softmax normalizers (redundant per lane; done once) ----
  const float* __restrict__ wrow = weight + h * KK;
  float wmax = -3.4e38f;
  #pragma unroll
  for (int k = 0; k < KK; ++k) wmax = fmaxf(wmax, wrow[k]);
  float wsum = 0.f;
  #pragma unroll
  for (int k = 0; k < KK; ++k) wsum += __expf(wrow[k] - wmax);
  const float winv = 1.0f / wsum;

  // ---- build Toeplitz weight bands A0/A1 in the wave32 A-matrix layout ----
  // A k order used here (contiguous, 16-bit A, 16x32): lane<16 -> M=lane, K halves {0..15};
  // lane>=16 -> M=lane-16, K halves {16..31}.
  const int i     = lane & 15;   // output-time row M
  const int kgrpA = lane >> 4;

  v16h a0, a1;
  #pragma unroll
  for (int hh = 0; hh < 16; ++hh) {
    const int v = hh >> 1, r = hh & 1;
    int Kidx = hh + (kgrpA ? 16 : 0);
    const int m0 = Kidx - i - 1;     // tap index fed by B0
    const int m1 = Kidx + 31 - i;    // tap index fed by B1
    float w0 = (m0 >= 0 && m0 < KK) ? __expf(wrow[m0] - wmax) * winv : 0.f;
    float w1 = (m1 < KK)            ? __expf(wrow[m1] - wmax) * winv : 0.f;
    a0[hh] = (_Float16)w0;
    a1[hh] = (_Float16)w1;
  }

  // ---- B / D lane mapping ----
  // B k order used here (contiguous, 16-bit B, 32x16): lane = N + 16*Kgroup; halves hh -> K = 16*Kgroup + hh.
  // D layout (f32 16x16): lane = N + 16*g; VGPR v -> M = v + 8*g.
  const int n    = lane & 15;
  const int kgrp = lane >> 4;
  const size_t chOff = (size_t)b * CTT + (size_t)(ch0 + n) * TT;
  const float* __restrict__ p = inp + chOff;
  float*       __restrict__ q = out + chOff;

  v8f cinit;
  {
    const float bv = bias[h];
    #pragma unroll
    for (int v = 0; v < 8; ++v) cinit[v] = bv;
  }

  const int t0base = tchunk * 256;

  for (int tt = 0; tt < 16; ++tt) {
    const int t0 = t0base + tt * 16;
    const int s0 = t0 - 16 + 16 * kgrp;   // this lane's B0 run (16 f32, 16-aligned)
    const int s1 = t0 + 16 + 16 * kgrp;   // this lane's B1 run

    float x0[16], x1[16];
    if (t0 >= 16 && t0 + 48 <= TT) {
      // fast interior path: 8x aligned global_load_b128 per lane
      const float4* __restrict__ p40 = (const float4*)(p + s0);
      const float4* __restrict__ p41 = (const float4*)(p + s1);
      #pragma unroll
      for (int u = 0; u < 4; ++u) {
        const float4 v0 = p40[u];
        const float4 v1 = p41[u];
        x0[4 * u + 0] = v0.x; x0[4 * u + 1] = v0.y;
        x0[4 * u + 2] = v0.z; x0[4 * u + 3] = v0.w;
        x1[4 * u + 0] = v1.x; x1[4 * u + 1] = v1.y;
        x1[4 * u + 2] = v1.z; x1[4 * u + 3] = v1.w;
      }
    } else {
      // boundary tiles: guarded scalar loads with zero padding
      #pragma unroll
      for (int u = 0; u < 16; ++u) {
        const int ta = s0 + u, tb = s1 + u;
        x0[u] = (ta >= 0 && ta < TT) ? p[ta] : 0.f;
        x1[u] = (tb >= 0 && tb < TT) ? p[tb] : 0.f;
      }
    }

    // f32 -> f16 operands (bandwidth stays f32; compute goes to the matrix pipe)
    v16h b0, b1;
    #pragma unroll
    for (int u = 0; u < 16; ++u) {
      b0[u] = (_Float16)x0[u];
      b1[u] = (_Float16)x1[u];
    }

    // D = A1*B1 + (A0*B0 + (bias))  via two chained WMMAs, f32 accumulation
    v8f acc = __builtin_amdgcn_wmma_f32_16x16x32_f16(
        false, a0, false, b0, (short)0, cinit, false, false);
    acc = __builtin_amdgcn_wmma_f32_16x16x32_f16(
        false, a1, false, b1, (short)0, acc, false, false);

    // store 8 consecutive output times per lane: two aligned global_store_b128
    {
      float4 o0, o1;
      o0.x = acc[0]; o0.y = acc[1]; o0.z = acc[2]; o0.w = acc[3];
      o1.x = acc[4]; o1.y = acc[5]; o1.z = acc[6]; o1.w = acc[7];
      float4* __restrict__ q4 = (float4*)(q + t0 + 8 * kgrp);
      q4[0] = o0;
      q4[1] = o1;
    }

    // speculative prefetch of the next window's fresh data
    if (s1 + 31 < TT) __builtin_prefetch(p + s1 + 16, 0, 1);
  }
}

extern "C" void kernel_launch(void* const* d_in, const int* in_sizes, int n_in,
                              void* d_out, int out_size, void* d_ws, size_t ws_size,
                              hipStream_t stream) {
  (void)in_sizes; (void)n_in; (void)out_size; (void)d_ws; (void)ws_size;
  const float* inp    = (const float*)d_in[0];
  const float* weight = (const float*)d_in[1];
  const float* bias   = (const float*)d_in[2];
  float* out          = (float*)d_out;

  // 8192 waves total: (8 b * 16 h * 4 chblk) channel-tiles x 16 T-chunks.
  // 256 threads = 8 waves per block -> 1024 blocks.
  dim3 grid(1024), block(256);
  lconv_wmma_kernel<<<grid, block, 0, stream>>>(inp, weight, bias, out);
}
